// MambaBlock_50208167690245
// MI455X (gfx1250) — matrix operand order repaired and checked by reading, not compiled, not run on hardware
//
#include <hip/hip_runtime.h>
#include <hip/hip_bf16.h>
#include <math.h>

typedef __attribute__((ext_vector_type(16))) _Float16 v16h;
typedef __attribute__((ext_vector_type(8)))  _Float16 v8h;
typedef __attribute__((ext_vector_type(8)))  float    v8f;

#define DIM   192
#define STATE 64
#define BATCH 8
#define HW    4096          // 64*64
#define NTOK  (BATCH*HW)    // 32768
#define NCHUNK 64
#define CLEN   64           // NCHUNK*CLEN == HW

// ---------- WMMA fragment helpers (16x16x32 f16 -> f32) ----------

// A-matrix 16x32 f16 from LDS tile (row stride ld halves).
// lanes 0-15: M=lane, K = kk+0..7 then kk+16..23
// lanes 16-31: M=lane-16, K = kk+8..15 then kk+24..31
__device__ inline v16h ldsA16x32(const _Float16* base, int ld, int m0, int kk) {
  int lane = threadIdx.x & 31;
  int row  = m0 + (lane & 15);
  int off  = (lane >> 4) * 8;
  const _Float16* p0 = base + row * ld + kk + off;
  v8h lo = *(const v8h*)p0;         // 16B aligned (strides/offsets multiples of 8 halves)
  v8h hi = *(const v8h*)(p0 + 16);
  v16h a;
#pragma unroll
  for (int i = 0; i < 8; ++i) { a[i] = lo[i]; a[i + 8] = hi[i]; }
  return a;
}

// B-matrix 32x16 f16 from a pre-converted global row-major [out,in] f16 weight (ld = in).
__device__ inline v16h gblB32x16h(const _Float16* __restrict__ W, int ld, int n0, int kk) {
  int lane = threadIdx.x & 31;
  int n = n0 + (lane & 15);
  int k = kk + ((lane >> 4) * 8);
  const v8h* p = (const v8h*)(W + (size_t)n * ld + k);
  v8h lo = p[0];
  v8h hi = p[2];
  v16h b;
#pragma unroll
  for (int i = 0; i < 8; ++i) { b[i] = lo[i]; b[i + 8] = hi[i]; }
  return b;
}

__device__ inline v8f wmma_f16(v16h a, v16h b, v8f c) {
  return __builtin_amdgcn_wmma_f32_16x16x32_f16(false, a, false, b, (short)0, c, false, false);
}

__device__ inline float softplusf(float x) {
  return x > 20.f ? x : log1pf(expf(x));
}

// ---------- Weight f32 -> f16 conversion (one-time, tiny) ----------
__global__ __launch_bounds__(256)
void kw_cvt(const float* __restrict__ src, _Float16* __restrict__ dst, int n) {
  int i = blockIdx.x * 256 + threadIdx.x;
  if (i < n) dst[i] = (_Float16)src[i];
}

// ---------- Kernel A: LN1 + ssm-LN + fused 192->256 projection GEMM + SSM pointwise ----------
__global__ __launch_bounds__(128)
void ka_proj(const float* __restrict__ x,
             const float* __restrict__ ln1_g, const float* __restrict__ ln1_b,
             const float* __restrict__ sg,    const float* __restrict__ sb,
             const _Float16* __restrict__ wcomb,
             const float* __restrict__ dt_b,  const float* __restrict__ xp_b,
             const float* __restrict__ A_log,
             float* __restrict__ A_arr, float* __restrict__ BX_arr,
             float* __restrict__ C_arr) {
  __shared__ __align__(16) char smem[32768 + 32 * 208 * 2];
  float*    xs = (float*)smem;                 // [32][200]  (aliased below)
  float*    ob = (float*)smem;                 // [32][256]  raw GEMM out (alias of xs)
  _Float16* xn = (_Float16*)(smem + 32768);    // [32][208]

  const int t0 = blockIdx.x * 32;
  for (int idx = threadIdx.x; idx < 32 * DIM; idx += 128) {
    int i = idx / DIM, c = idx % DIM;
    int t = t0 + i, b = t >> 12, hw = t & (HW - 1);
    xs[i * 200 + c] = x[((size_t)b * DIM + c) * HW + hw];
  }
  __syncthreads();

  {
    int i = threadIdx.x >> 2;           // token 0..31
    int q = threadIdx.x & 3;
    int c0 = q * 48;
    float s = 0.f, ss = 0.f;
    for (int c = c0; c < c0 + 48; ++c) { float v = xs[i * 200 + c]; s += v; ss += v * v; }
    s  += __shfl_xor(s, 1, 4);  ss += __shfl_xor(ss, 1, 4);
    s  += __shfl_xor(s, 2, 4);  ss += __shfl_xor(ss, 2, 4);
    float m1 = s / DIM;
    float r1 = rsqrtf(ss / DIM - m1 * m1 + 1e-5f);
    s = 0.f; ss = 0.f;
    for (int c = c0; c < c0 + 48; ++c) {
      float v = (xs[i * 200 + c] - m1) * r1 * ln1_g[c] + ln1_b[c];
      s += v; ss += v * v;
    }
    s  += __shfl_xor(s, 1, 4);  ss += __shfl_xor(ss, 1, 4);
    s  += __shfl_xor(s, 2, 4);  ss += __shfl_xor(ss, 2, 4);
    float m2 = s / DIM;
    float r2 = rsqrtf(ss / DIM - m2 * m2 + 1e-5f);
    for (int c = c0; c < c0 + 48; ++c) {
      float v = (xs[i * 200 + c] - m1) * r1 * ln1_g[c] + ln1_b[c];
      xn[i * 208 + c] = (_Float16)((v - m2) * r2 * sg[c] + sb[c]);
    }
  }
  __syncthreads();

  const int wave = threadIdx.x >> 5, lane = threadIdx.x & 31;
  const int msub = wave & 1;
  const int jb   = (wave >> 1) * 8;
  v16h af[6];
#pragma unroll
  for (int k = 0; k < 6; ++k) af[k] = ldsA16x32(xn, 208, msub * 16, k * 32);
  for (int jj = 0; jj < 8; ++jj) {
    int j = jb + jj;
    v8f acc = {};
#pragma unroll
    for (int k = 0; k < 6; ++k) {
      v16h bf = gblB32x16h(wcomb, DIM, j * 16, k * 32);
      acc = wmma_f16(af[k], bf, acc);
    }
    int nc   = j * 16 + (lane & 15);
    int mrow = msub * 16 + ((lane >> 4) * 8);
#pragma unroll
    for (int r = 0; r < 8; ++r) ob[(mrow + r) * 256 + nc] = acc[r];
  }
  __syncthreads();

  for (int idx = threadIdx.x; idx < 32 * STATE; idx += 128) {
    int i = idx >> 6, s = idx & 63;
    float dt_raw = ob[i * 256 + s] + dt_b[s];
    float delta  = softplusf(dt_raw) * 0.01f + 1e-4f;
    float A      = -expf(A_log[s]);
    float dA     = fminf(fmaxf(delta * A, -10.f), -1e-4f);
    float Ad     = fminf(fmaxf(expf(dA), 0.001f), 0.999f);
    float Bt     = ob[i * 256 + 64 + s];
    float Ct     = ob[i * 256 + 128 + s];
    float xp     = ob[i * 256 + 192 + s] + xp_b[s];
    size_t o = (size_t)(t0 + i) * STATE + s;
    A_arr[o]  = Ad;
    BX_arr[o] = delta * Bt * xp;
    C_arr[o]  = Ct;
  }
}

// ---------- Scan (3-phase chunked) ----------
__global__ __launch_bounds__(256)
void kb1_reduce(const float* __restrict__ A_arr, const float* __restrict__ BX,
                float* __restrict__ cA, float* __restrict__ cB) {
  int gid = blockIdx.x * 256 + threadIdx.x;
  if (gid >= BATCH * NCHUNK * STATE) return;
  int s = gid & 63, chunk = (gid >> 6) & (NCHUNK - 1), b = gid >> 12;
  size_t base = ((size_t)b * HW + chunk * CLEN) * STATE + s;
  float aP = 1.f, bx = 0.f;
  for (int t = 0; t < CLEN; ++t) {
    size_t i = base + (size_t)t * STATE;
    float a = A_arr[i];
    aP = a * aP;
    bx = a * bx + BX[i];
  }
  cA[gid] = aP;
  cB[gid] = bx;
}

__global__ __launch_bounds__(256)
void kb2_chain(const float* __restrict__ cA, const float* __restrict__ cB,
               float* __restrict__ hin) {
  int gid = blockIdx.x * 256 + threadIdx.x;
  if (gid >= BATCH * STATE) return;
  int s = gid & 63, b = gid >> 6;
  float h = 0.f;
  for (int chunk = 0; chunk < NCHUNK; ++chunk) {
    size_t i = ((size_t)b * NCHUNK + chunk) * STATE + s;
    hin[i] = h;
    h = cA[i] * h + cB[i];
  }
}

__global__ __launch_bounds__(256)
void kb3_emit(const float* __restrict__ A_arr, const float* __restrict__ BX,
              const float* __restrict__ C_arr, const float* __restrict__ hin,
              float* __restrict__ y_pre) {
  int gid = blockIdx.x * 256 + threadIdx.x;
  if (gid >= BATCH * NCHUNK * STATE) return;
  int s = gid & 63, chunk = (gid >> 6) & (NCHUNK - 1), b = gid >> 12;
  size_t base = ((size_t)b * HW + chunk * CLEN) * STATE + s;
  float h = hin[gid];
  for (int t = 0; t < CLEN; ++t) {
    size_t i = base + (size_t)t * STATE;
    h = A_arr[i] * h + BX[i];
    y_pre[i] = C_arr[i] * h;
  }
}

// ---------- Depthwise 3x3 conv + BatchNorm (eval) ----------
__global__ __launch_bounds__(256)
void kl_conv(const float* __restrict__ x, const float* __restrict__ dw,
             const float* __restrict__ bn_g, const float* __restrict__ bn_b,
             const float* __restrict__ rm, const float* __restrict__ rv,
             float* __restrict__ xloc) {
  int idx = blockIdx.x * 256 + threadIdx.x;
  if (idx >= NTOK * DIM) return;
  int hw = idx & (HW - 1);
  int bc = idx >> 12;
  int c  = bc % DIM;
  int h = hw >> 6, w = hw & 63;
  const float* xp = x + (size_t)bc * HW;
  const float* wp = dw + c * 9;
  float acc = 0.f;
#pragma unroll
  for (int kh = 0; kh < 3; ++kh) {
    int hh = h + kh - 1;
    if (hh < 0 || hh >= 64) continue;
#pragma unroll
    for (int kw = 0; kw < 3; ++kw) {
      int ww = w + kw - 1;
      if (ww < 0 || ww >= 64) continue;
      acc += xp[hh * 64 + ww] * wp[kh * 3 + kw];
    }
  }
  float sc = bn_g[c] * rsqrtf(rv[c] + 1e-5f);
  xloc[idx] = (acc - rm[c]) * sc + bn_b[c];
}

// ---------- Kernel C: out-projection (64->192) + D-term + residual merge ----------
__global__ __launch_bounds__(128)
void kc_combine(const float* __restrict__ x,
                const float* __restrict__ ln1_g, const float* __restrict__ ln1_b,
                const float* __restrict__ y_pre, const _Float16* __restrict__ wout_h,
                const float* __restrict__ out_b, const float* __restrict__ Dp,
                const float* __restrict__ xloc,
                const float* __restrict__ a_loc, const float* __restrict__ a_ssm,
                float* __restrict__ xr) {
  __shared__ __align__(16) _Float16 yt[64 * 72];
  __shared__ float sm1[64], sr1[64];
  const int t0 = blockIdx.x * 64;

  for (int idx = threadIdx.x; idx < 64 * STATE; idx += 128) {
    int i = idx >> 6, s = idx & 63;
    yt[i * 72 + s] = (_Float16)y_pre[(size_t)(t0 + i) * STATE + s];
  }
  {
    int i = threadIdx.x >> 1;
    int half = threadIdx.x & 1;
    int t = t0 + i, b = t >> 12, hw = t & (HW - 1);
    const float* xp = x + (size_t)b * DIM * HW + hw;
    float s = 0.f, ss = 0.f;
    for (int c = half * 96; c < half * 96 + 96; ++c) {
      float v = xp[(size_t)c * HW]; s += v; ss += v * v;
    }
    s  += __shfl_xor(s, 1, 2);
    ss += __shfl_xor(ss, 1, 2);
    if (half == 0) {
      float m = s / DIM;
      sm1[i] = m;
      sr1[i] = rsqrtf(ss / DIM - m * m + 1e-5f);
    }
  }
  __syncthreads();

  const int wave = threadIdx.x >> 5, lane = threadIdx.x & 31;
  const float al = a_loc[0], as_ = a_ssm[0];
  v16h af[2];
#pragma unroll
  for (int k = 0; k < 2; ++k) af[k] = ldsA16x32(yt, 72, wave * 16, k * 32);
  for (int j = 0; j < 12; ++j) {
    v8f acc = {};
#pragma unroll
    for (int k = 0; k < 2; ++k) {
      v16h bf = gblB32x16h(wout_h, STATE, j * 16, k * 32);
      acc = wmma_f16(af[k], bf, acc);
    }
    int c    = j * 16 + (lane & 15);
    int mrow = wave * 16 + ((lane >> 4) * 8);
    float ob_c = out_b[c];
    float dcl  = fminf(fmaxf(Dp[c], -2.f), 2.f);
    float g1 = ln1_g[c], b1 = ln1_b[c];
#pragma unroll
    for (int r = 0; r < 8; ++r) {
      int i = mrow + r, t = t0 + i, b = t >> 12, hw = t & (HW - 1);
      size_t gi = ((size_t)b * DIM + c) * HW + hw;
      float xv   = x[gi];
      float xln1 = (xv - sm1[i]) * sr1[i] * g1 + b1;
      float ssm  = acc[r] + ob_c + dcl * xln1;
      xr[gi] = xv + al * xloc[gi] + as_ * ssm;
    }
  }
}

// ---------- Kernel D: LN2 + MLP (192->768 GELU 768->192) + final residual ----------
__global__ __launch_bounds__(128)
void kd_mlp(const float* __restrict__ xr,
            const float* __restrict__ ln2_g, const float* __restrict__ ln2_b,
            const _Float16* __restrict__ w1h, const float* __restrict__ b1,
            const _Float16* __restrict__ w2h, const float* __restrict__ b2,
            const float* __restrict__ a_mlp, float* __restrict__ out) {
  __shared__ __align__(16) char smem[49664 + 32 * 208 * 2];
  float*    xs  = (float*)smem;                 // [32][200] (aliased by hid)
  _Float16* hid = (_Float16*)smem;              // [32][776]
  _Float16* xn  = (_Float16*)(smem + 49664);    // [32][208]

  const int t0 = blockIdx.x * 32;
  __builtin_prefetch(w1h, 0, 1);
  __builtin_prefetch(w2h, 0, 1);
  for (int idx = threadIdx.x; idx < 32 * DIM; idx += 128) {
    int i = idx / DIM, c = idx % DIM;
    int t = t0 + i, b = t >> 12, hw = t & (HW - 1);
    xs[i * 200 + c] = xr[((size_t)b * DIM + c) * HW + hw];
  }
  __syncthreads();
  {
    int i = threadIdx.x >> 2;
    int q = threadIdx.x & 3;
    int c0 = q * 48;
    float s = 0.f, ss = 0.f;
    for (int c = c0; c < c0 + 48; ++c) { float v = xs[i * 200 + c]; s += v; ss += v * v; }
    s  += __shfl_xor(s, 1, 4);  ss += __shfl_xor(ss, 1, 4);
    s  += __shfl_xor(s, 2, 4);  ss += __shfl_xor(ss, 2, 4);
    float m = s / DIM, r = rsqrtf(ss / DIM - m * m + 1e-5f);
    for (int c = c0; c < c0 + 48; ++c)
      xn[i * 208 + c] = (_Float16)((xs[i * 200 + c] - m) * r * ln2_g[c] + ln2_b[c]);
  }
  __syncthreads();

  const int wave = threadIdx.x >> 5, lane = threadIdx.x & 31;
  const int msub = wave & 1;

  v16h af[6];
#pragma unroll
  for (int k = 0; k < 6; ++k) af[k] = ldsA16x32(xn, 208, msub * 16, k * 32);
  const int jb1 = (wave >> 1) * 24;
  for (int jj = 0; jj < 24; ++jj) {
    int j = jb1 + jj;
    v8f acc = {};
#pragma unroll
    for (int k = 0; k < 6; ++k) {
      v16h bf = gblB32x16h(w1h, DIM, j * 16, k * 32);
      acc = wmma_f16(af[k], bf, acc);
    }
    int n    = j * 16 + (lane & 15);
    int mrow = msub * 16 + ((lane >> 4) * 8);
    float bb = b1[n];
#pragma unroll
    for (int r = 0; r < 8; ++r) {
      float v = acc[r] + bb;
      float g = 0.5f * v * (1.f + erff(v * 0.70710678f));
      hid[(mrow + r) * 776 + n] = (_Float16)g;
    }
  }
  __syncthreads();

  const float am = a_mlp[0];
  const int jb2 = (wave >> 1) * 6;
  for (int jj = 0; jj < 6; ++jj) {
    int j = jb2 + jj;
    v8f acc = {};
    for (int k = 0; k < 24; ++k) {
      v16h a  = ldsA16x32(hid, 776, msub * 16, k * 32);
      v16h bf = gblB32x16h(w2h, 768, j * 16, k * 32);
      acc = wmma_f16(a, bf, acc);
    }
    int c    = j * 16 + (lane & 15);
    int mrow = msub * 16 + ((lane >> 4) * 8);
    float bb = b2[c];
#pragma unroll
    for (int r = 0; r < 8; ++r) {
      int i = mrow + r, t = t0 + i, b = t >> 12, hw = t & (HW - 1);
      size_t gi = ((size_t)b * DIM + c) * HW + hw;
      out[gi] = xr[gi] + am * (acc[r] + bb);
    }
  }
}

// ---------- launch ----------
extern "C" void kernel_launch(void* const* d_in, const int* in_sizes, int n_in,
                              void* d_out, int out_size, void* d_ws, size_t ws_size,
                              hipStream_t stream) {
  (void)in_sizes; (void)n_in; (void)out_size; (void)ws_size;
  const float* x     = (const float*)d_in[0];
  const float* ln1_g = (const float*)d_in[1];
  const float* ln1_b = (const float*)d_in[2];
  const float* ln2_g = (const float*)d_in[3];
  const float* ln2_b = (const float*)d_in[4];
  const float* dw_w  = (const float*)d_in[5];
  const float* bn_g  = (const float*)d_in[6];
  const float* bn_b  = (const float*)d_in[7];
  const float* bn_rm = (const float*)d_in[8];
  const float* bn_rv = (const float*)d_in[9];
  const float* sg    = (const float*)d_in[10];
  const float* sb    = (const float*)d_in[11];
  const float* xp_w  = (const float*)d_in[12];
  const float* xp_b  = (const float*)d_in[13];
  const float* dt_w  = (const float*)d_in[14];
  const float* dt_b  = (const float*)d_in[15];
  const float* A_log = (const float*)d_in[16];
  const float* B_w   = (const float*)d_in[17];
  const float* C_w   = (const float*)d_in[18];
  const float* Dp    = (const float*)d_in[19];
  const float* out_w = (const float*)d_in[20];
  const float* out_b = (const float*)d_in[21];
  const float* w1    = (const float*)d_in[22];
  const float* b1    = (const float*)d_in[23];
  const float* w2    = (const float*)d_in[24];
  const float* b2    = (const float*)d_in[25];
  const float* a_loc = (const float*)d_in[26];
  const float* a_ssm = (const float*)d_in[27];
  const float* a_mlp = (const float*)d_in[28];

  char* ws = (char*)d_ws;
  const size_t szS = (size_t)NTOK * STATE * sizeof(float);              // 8 MB
  const size_t szD = (size_t)NTOK * DIM * sizeof(float);                // 24 MB
  const size_t szC = (size_t)BATCH * NCHUNK * STATE * sizeof(float);    // 128 KB
  float* A_arr = (float*)(ws);
  float* BXa   = (float*)(ws + szS);
  float* Ca    = (float*)(ws + 2 * szS);
  float* ypre  = (float*)(ws + 3 * szS);
  float* xloc  = (float*)(ws + 4 * szS);
  float* xr    = (float*)(ws + 4 * szS + szD);
  float* cA    = (float*)(ws + 4 * szS + 2 * szD);
  float* cB    = (float*)(ws + 4 * szS + 2 * szD + szC);
  float* hin   = (float*)(ws + 4 * szS + 2 * szD + 2 * szC);
  char*  wsf16 = ws + 4 * szS + 2 * szD + 3 * szC;
  _Float16* wcomb  = (_Float16*)wsf16;                    // [256][192] = dt|B|C|xp
  _Float16* wout_h = wcomb + 256 * 192;                   // [192][64]
  _Float16* w1_h   = wout_h + 192 * 64;                   // [768][192]
  _Float16* w2_h   = w1_h + 768 * 192;                    // [192][768]

  kw_cvt<<<(64 * 192 + 255) / 256, 256, 0, stream>>>(dt_w, wcomb, 64 * 192);
  kw_cvt<<<(64 * 192 + 255) / 256, 256, 0, stream>>>(B_w, wcomb + 64 * 192, 64 * 192);
  kw_cvt<<<(64 * 192 + 255) / 256, 256, 0, stream>>>(C_w, wcomb + 128 * 192, 64 * 192);
  kw_cvt<<<(64 * 192 + 255) / 256, 256, 0, stream>>>(xp_w, wcomb + 192 * 192, 64 * 192);
  kw_cvt<<<(192 * 64 + 255) / 256, 256, 0, stream>>>(out_w, wout_h, 192 * 64);
  kw_cvt<<<(768 * 192 + 255) / 256, 256, 0, stream>>>(w1, w1_h, 768 * 192);
  kw_cvt<<<(192 * 768 + 255) / 256, 256, 0, stream>>>(w2, w2_h, 192 * 768);

  ka_proj<<<NTOK / 32, 128, 0, stream>>>(x, ln1_g, ln1_b, sg, sb, wcomb,
                                         dt_b, xp_b, A_log, A_arr, BXa, Ca);
  kb1_reduce<<<(BATCH * NCHUNK * STATE) / 256, 256, 0, stream>>>(A_arr, BXa, cA, cB);
  kb2_chain<<<2, 256, 0, stream>>>(cA, cB, hin);
  kb3_emit<<<(BATCH * NCHUNK * STATE) / 256, 256, 0, stream>>>(A_arr, BXa, Ca, hin, ypre);
  kl_conv<<<(NTOK * DIM + 255) / 256, 256, 0, stream>>>(x, dw_w, bn_g, bn_b, bn_rm, bn_rv, xloc);
  kc_combine<<<NTOK / 64, 128, 0, stream>>>(x, ln1_g, ln1_b, ypre, wout_h, out_b,
                                            Dp, xloc, a_loc, a_ssm, xr);
  kd_mlp<<<NTOK / 32, 128, 0, stream>>>(xr, ln2_g, ln2_b, w1_h, b1, w2_h, b2, a_mlp,
                                        (float*)d_out);
}
